// ModularAttentionDev_23854248362901
// MI455X (gfx1250) — matrix operand order repaired and checked by reading, not compiled, not run on hardware
//
#include <hip/hip_runtime.h>

typedef __attribute__((ext_vector_type(16))) __bf16 v16bf;
typedef __attribute__((ext_vector_type(8)))  float  v8f;
typedef __attribute__((ext_vector_type(2)))  float  v2f;
typedef __attribute__((ext_vector_type(4)))  unsigned int u32x4;
typedef __attribute__((ext_vector_type(8)))  int    i32x8;
typedef __attribute__((ext_vector_type(4)))  int    i32x4;

#define BATCH   8
#define SEQ     8192
#define DM      256   // D_MODEL

// ---------------------------------------------------------------------------
// Kernel 1: scores[b] = (1/16) * K^T V     (M=256 d, N=256 e, K=S=8192)
// Block = 128 threads (4 waves), 64x64 tile. bf16 WMMA, fp32 accumulate.
// Global loads are float4 (b128); fragments staged in frag-order LDS so each
// lane's WMMA operand is one contiguous 32B LDS read.
// ---------------------------------------------------------------------------
__global__ __launch_bounds__(128) void la_scores_kernel(
    const float* __restrict__ K, const float* __restrict__ V,
    float* __restrict__ scores) {
  __shared__ __align__(32) __bf16 ldsA[4 * 32 * 16];  // [wave][lane][16]
  __shared__ __align__(32) __bf16 ldsB[4 * 32 * 16];  // [tile][lane][16]

  const int b    = blockIdx.x >> 4;
  const int tile = blockIdx.x & 15;
  const int d0 = (tile >> 2) * 64;
  const int e0 = (tile & 3) * 64;
  const int tid = threadIdx.x;
  const int w = tid >> 5;
  const int l = tid & 31;

  const float* kb = K + (size_t)b * SEQ * DM;
  const float* vb = V + (size_t)b * SEQ * DM;

  v8f acc0 = {}, acc1 = {}, acc2 = {}, acc3 = {};

  for (int s0 = 0; s0 < SEQ; s0 += 32) {
    // --- stage 32(s) x 64(d/e) chunks of k and v (float4 loads) ---
    #pragma unroll
    for (int it = 0; it < 4; ++it) {
      const int iv = tid + it * 128;       // 0..511 float4 index
      const int sl = iv >> 4;              // 0..31  (WMMA K index)
      const int dl = (iv & 15) << 2;       // 0,4,...,60
      const float4 kv = *(const float4*)&kb[(size_t)(s0 + sl) * DM + d0 + dl];
      const float4 vv = *(const float4*)&vb[(size_t)(s0 + sl) * DM + e0 + dl];
      if (s0 + 32 < SEQ) {  // prefetch next streaming chunk
        __builtin_prefetch(&kb[(size_t)(s0 + 32 + sl) * DM + d0 + dl], 0, 3);
        __builtin_prefetch(&vb[(size_t)(s0 + 32 + sl) * DM + e0 + dl], 0, 3);
      }
      const int g  = dl >> 4;              // 16-wide sub-block (wave / tile id)
      const int ha = (sl >> 3) & 1;
      const int ja = (sl & 7) + 8 * (sl >> 4);
      const int hb = (sl >> 3) & 1;
      const int jb = (sl & 7) + 8 * (sl >> 4);
      const float kf[4] = {kv.x, kv.y, kv.z, kv.w};
      const float vf[4] = {vv.x, vv.y, vv.z, vv.w};
      #pragma unroll
      for (int u = 0; u < 4; ++u) {
        const int m = (dl + u) & 15;
        ldsA[((g * 32 + (m + 16 * ha)) << 4) + ja] = (__bf16)kf[u];
        ldsB[((g * 32 + (m + 16 * hb)) << 4) + jb] = (__bf16)vf[u];
      }
    }
    __syncthreads();

    const v16bf aF = *(const v16bf*)&ldsA[(w * 32 + l) << 4];
    const v16bf b0 = *(const v16bf*)&ldsB[(0 * 32 + l) << 4];
    const v16bf b1 = *(const v16bf*)&ldsB[(1 * 32 + l) << 4];
    const v16bf b2 = *(const v16bf*)&ldsB[(2 * 32 + l) << 4];
    const v16bf b3 = *(const v16bf*)&ldsB[(3 * 32 + l) << 4];
    acc0 = __builtin_amdgcn_wmma_f32_16x16x32_bf16(false, aF, false, b0, (short)0, acc0, false, false);
    acc1 = __builtin_amdgcn_wmma_f32_16x16x32_bf16(false, aF, false, b1, (short)0, acc1, false, false);
    acc2 = __builtin_amdgcn_wmma_f32_16x16x32_bf16(false, aF, false, b2, (short)0, acc2, false, false);
    acc3 = __builtin_amdgcn_wmma_f32_16x16x32_bf16(false, aF, false, b3, (short)0, acc3, false, false);
    __syncthreads();
  }

  const float scale = 0.0625f;  // 1/sqrt(256)
  float* sb = scores + (size_t)b * DM * DM;
  const int h = l >> 4, n = l & 15;
  v8f accs[4] = {acc0, acc1, acc2, acc3};
  #pragma unroll
  for (int tt = 0; tt < 4; ++tt) {
    #pragma unroll
    for (int r = 0; r < 8; ++r) {
      sb[(d0 + 16 * w + r + 8 * h) * DM + (e0 + 16 * tt + n)] = accs[tt][r] * scale;
    }
  }
}

// ---------------------------------------------------------------------------
// Kernel 2a/2b: denom[b,d] = 1 + sum_s exp(q)  (deterministic two-stage)
// ---------------------------------------------------------------------------
__global__ __launch_bounds__(256) void la_colsum_exp_kernel(
    const float* __restrict__ Q, float* __restrict__ partials) {
  const int b = blockIdx.x >> 5, chunk = blockIdx.x & 31, d = threadIdx.x;
  const float* qb = Q + (size_t)b * SEQ * DM + (size_t)chunk * 256 * DM;
  float s = 0.f;
  for (int i = 0; i < 256; ++i) s += __expf(qb[(size_t)i * DM + d]);
  partials[(size_t)blockIdx.x * DM + d] = s;
}

__global__ __launch_bounds__(256) void la_reduce_denom_kernel(
    const float* __restrict__ partials, float* __restrict__ denom) {
  const int b = blockIdx.x, d = threadIdx.x;
  float s = 1.0f;  // the "+1" of the reference denominator
  for (int c = 0; c < 32; ++c) s += partials[(size_t)(b * 32 + c) * DM + d];
  denom[b * DM + d] = s;
}

// ---------------------------------------------------------------------------
// Kernel 3a/3b: sm[b,d] = sum_s exp(exp(q)/denom)   (softmax normalizer)
// ---------------------------------------------------------------------------
__global__ __launch_bounds__(256) void la_colsum_sm_kernel(
    const float* __restrict__ Q, const float* __restrict__ denom,
    float* __restrict__ partials) {
  const int b = blockIdx.x >> 5, chunk = blockIdx.x & 31, d = threadIdx.x;
  const float invden = 1.0f / denom[b * DM + d];
  const float* qb = Q + (size_t)b * SEQ * DM + (size_t)chunk * 256 * DM;
  float s = 0.f;
  for (int i = 0; i < 256; ++i)
    s += __expf(__expf(qb[(size_t)i * DM + d]) * invden);
  partials[(size_t)blockIdx.x * DM + d] = s;
}

__global__ __launch_bounds__(256) void la_reduce_sm_kernel(
    const float* __restrict__ partials, float* __restrict__ sm) {
  const int b = blockIdx.x, d = threadIdx.x;
  float s = 0.f;
  for (int c = 0; c < 32; ++c) s += partials[(size_t)(b * 32 + c) * DM + d];
  sm[b * DM + d] = s;
}

// ---------------------------------------------------------------------------
// Kernel 4: out[b] = W @ scores[b], W[s,d] = exp(exp(q)/denom[d]) / sm[d]
// M=8192, N=256, K=256. Full-precision f32 WMMA (16x16x4).
// B (scores) is DMA'd into LDS with the Tensor Data Mover in two 128x64 f32
// half-slabs (32 KB each) -- pure copy, no VGPR traffic -- tracked with
// TENSORcnt. A is computed from q (exp transform) and staged in frag order.
// ---------------------------------------------------------------------------
__global__ __launch_bounds__(128) void la_out_kernel(
    const float* __restrict__ Q, const float* __restrict__ scores,
    const float* __restrict__ denom, const float* __restrict__ sm,
    float* __restrict__ out) {
  __shared__ __align__(16) float slabB[128 * 64];       // 32 KB TDM target
  __shared__ __align__(16) float ldsA[4 * 32 * 8 * 2];  // 8 KB frag-order A
  __shared__ float sInvD[DM];
  __shared__ float sInvS[DM];

  const int et = blockIdx.x & 3;
  const int st = (blockIdx.x >> 2) & 127;
  const int b  = blockIdx.x >> 9;
  const int s0 = st * 64;
  const int e0 = et * 64;
  const int tid = threadIdx.x;
  const int w = tid >> 5;
  const int l = tid & 31;

  for (int i = tid; i < DM; i += 128) {
    sInvD[i] = 1.0f / denom[b * DM + i];
    sInvS[i] = 1.0f / sm[b * DM + i];
  }

  const float* qb = Q + (size_t)b * SEQ * DM;
  const float* sc = scores + (size_t)b * DM * DM;
  const unsigned lds_slab = (unsigned)(uintptr_t)&slabB[0];  // addr[31:0] = LDS offset

  v8f acc0 = {}, acc1 = {}, acc2 = {}, acc3 = {};

  for (int half = 0; half < 2; ++half) {
    // --- TDM: scores[128*half .. +128, e0 .. e0+64] (f32, row stride 256) ---
    if (tid < 32) {  // one wave issues the DMA; descriptor is block-uniform
      const unsigned long long ga =
          (unsigned long long)(uintptr_t)(sc + (size_t)(128 * half) * DM + e0);
      // D# group 0: count=1 | lds_addr | global_addr[56:0] | type=2
      const u32x4 g0 = {
          1u,
          lds_slab,
          (unsigned)(ga & 0xFFFFFFFFu),
          (unsigned)((ga >> 32) & 0x1FFFFFFu) | (2u << 30)};
      // D# group 1: data_size=4B; tensor_dim0/1=256; tile 64x128; stride0=256
      const i32x8 g1 = {
          (int)(2u << 16),                               // data_size code 2
          (int)((256u & 0xFFFFu) << 16),                 // tensor_dim0[15:0]
          (int)((256u >> 16) | ((256u & 0xFFFFu) << 16)),// td0[31:16] | td1[15:0]
          (int)((256u >> 16) | (64u << 16)),             // td1[31:16] | tile_dim0=64
          (int)128,                                      // tile_dim1=128, tile_dim2=0
          (int)256,                                      // tensor_dim0_stride[31:0]
          0,                                             // stride0[47:32] | stride1[15:0]
          0};                                            // stride1[47:16]
      const i32x4 gz4 = {0, 0, 0, 0};                    // dims 2..4 unused (2D)
      const i32x8 gz8 = {0, 0, 0, 0, 0, 0, 0, 0};
      __builtin_amdgcn_tensor_load_to_lds(g0, g1, gz4, gz4, gz8, 0);
      __builtin_amdgcn_s_wait_tensorcnt(0);
    }
    __syncthreads();  // publish slab (and sInvD/S on first pass)

    for (int kk = 0; kk < 128; kk += 32) {
      const int k0 = 128 * half + kk;
      // --- stage A chunk: 64 rows (s) x 32 cols (k), float4 loads, exp xform ---
      #pragma unroll
      for (int it = 0; it < 4; ++it) {
        const int iv = tid + it * 128;   // 0..511
        const int r = iv >> 3;           // 0..63 (s row)
        const int t = iv & 7;            // 0..7  (k-step; c = 4t)
        const int c = k0 + 4 * t;
        const float4 qv = *(const float4*)&qb[(size_t)(s0 + r) * DM + c];
        const float a0 = __expf(__expf(qv.x) * sInvD[c + 0]) * sInvS[c + 0];
        const float a1 = __expf(__expf(qv.y) * sInvD[c + 1]) * sInvS[c + 1];
        const float a2 = __expf(__expf(qv.z) * sInvD[c + 2]) * sInvS[c + 2];
        const float a3 = __expf(__expf(qv.w) * sInvD[c + 3]) * sInvS[c + 3];
        const int wa = r >> 4, m = r & 15;
        const v2f lo = {a0, a1};   // h=0 pair (k = 4t+0, 4t+1)
        const v2f hi = {a2, a3};   // h=1 pair (k = 4t+2, 4t+3)
        *(v2f*)&ldsA[((wa * 32 + m) * 8 + t) << 1]      = lo;
        *(v2f*)&ldsA[((wa * 32 + m + 16) * 8 + t) << 1] = hi;
      }
      __syncthreads();

      const int h = l >> 4, n = l & 15;
      #pragma unroll
      for (int t = 0; t < 8; ++t) {
        const v2f aF = *(const v2f*)&ldsA[((w * 32 + l) * 8 + t) << 1];
        const int kr = kk + 4 * t + 2 * h;   // slab row of frag element p=0
        const v2f bF0 = {slabB[kr * 64 +  0 + n], slabB[(kr + 1) * 64 +  0 + n]};
        const v2f bF1 = {slabB[kr * 64 + 16 + n], slabB[(kr + 1) * 64 + 16 + n]};
        const v2f bF2 = {slabB[kr * 64 + 32 + n], slabB[(kr + 1) * 64 + 32 + n]};
        const v2f bF3 = {slabB[kr * 64 + 48 + n], slabB[(kr + 1) * 64 + 48 + n]};
        acc0 = __builtin_amdgcn_wmma_f32_16x16x4_f32(false, aF, false, bF0, (short)0, acc0, false, false);
        acc1 = __builtin_amdgcn_wmma_f32_16x16x4_f32(false, aF, false, bF1, (short)0, acc1, false, false);
        acc2 = __builtin_amdgcn_wmma_f32_16x16x4_f32(false, aF, false, bF2, (short)0, acc2, false, false);
        acc3 = __builtin_amdgcn_wmma_f32_16x16x4_f32(false, aF, false, bF3, (short)0, acc3, false, false);
      }
      __syncthreads();
    }
  }

  float* ob = out + (size_t)b * SEQ * DM;
  const int h = l >> 4, n = l & 15;
  v8f accs[4] = {acc0, acc1, acc2, acc3};
  #pragma unroll
  for (int tt = 0; tt < 4; ++tt) {
    #pragma unroll
    for (int r = 0; r < 8; ++r) {
      ob[(size_t)(s0 + 16 * w + r + 8 * h) * DM + e0 + 16 * tt + n] = accs[tt][r];
    }
  }
}

// ---------------------------------------------------------------------------
// Host launcher. Workspace layout (floats):
//   [0 .. 524288)         scores   (B*256*256)
//   [524288 .. 526336)    denom    (B*256)
//   [526336 .. 528384)    sm       (B*256)
//   [528384 .. 593920)    partials (B*32*256), reused for both reductions
// ---------------------------------------------------------------------------
extern "C" void kernel_launch(void* const* d_in, const int* in_sizes, int n_in,
                              void* d_out, int out_size, void* d_ws, size_t ws_size,
                              hipStream_t stream) {
  const float* k = (const float*)d_in[0];
  const float* q = (const float*)d_in[1];
  const float* v = (const float*)d_in[2];
  float* out = (float*)d_out;

  float* ws       = (float*)d_ws;
  float* scores   = ws;
  float* denom    = scores + (size_t)BATCH * DM * DM;
  float* sm       = denom + BATCH * DM;
  float* partials = sm + BATCH * DM;

  la_scores_kernel<<<BATCH * 16, 128, 0, stream>>>(k, v, scores);
  la_colsum_exp_kernel<<<BATCH * 32, 256, 0, stream>>>(q, partials);
  la_reduce_denom_kernel<<<BATCH, 256, 0, stream>>>(partials, denom);
  la_colsum_sm_kernel<<<BATCH * 32, 256, 0, stream>>>(q, denom, partials);
  la_reduce_sm_kernel<<<BATCH, 256, 0, stream>>>(partials, sm);
  la_out_kernel<<<BATCH * 128 * 4, 128, 0, stream>>>(q, scores, denom, sm, out);
}
